// scnModel_64656437674378
// MI455X (gfx1250) — matrix operand order repaired and checked by reading, not compiled, not run on hardware
//
#include <hip/hip_runtime.h>

typedef __attribute__((ext_vector_type(16))) _Float16 v16h;
typedef __attribute__((ext_vector_type(8)))  _Float16 v8h;
typedef __attribute__((ext_vector_type(4)))  _Float16 v4h;
typedef __attribute__((ext_vector_type(2)))  _Float16 v2h;
typedef __attribute__((ext_vector_type(8)))  float    v8f;
typedef __attribute__((ext_vector_type(4)))  unsigned int v4u;
typedef __attribute__((ext_vector_type(8)))  int v8i;
typedef __attribute__((ext_vector_type(4)))  int v4i;

#define N_ROWS  100000
#define IN_DIM  168
#define KPAD    192      // 168 zero-padded to 6 chunks of 32
#define NCHUNK  6
#define ERR_DIM 24
#define DPAD    32       // 24 zero-padded to 32 (2 WMMA M-tiles)
#define CAND    256
#define NR      4
#define TM      32       // rows of X/E per workgroup

// LDS row strides in halves; multiples of 8 (16B) so b128 accesses stay aligned
#define XSTR 200         // x_lds:  [TM][XSTR]
#define WTSTR 40         // wT:     [CAND][WTSTR]  (32 K + 8 pad; TDM pad feature)
#define SSTR 40          // sT:     [CAND][SSTR]   (S^T, 32 rows + 8 pad)
#define ESTR 40          // e_lds:  [DPAD][ESTR]

// LDS layout (bytes)
#define OFF_X   0                    // 32*200*2 = 12800
#define OFF_E   12800                // 32*40*2  =  2560
#define OFF_W0  15360                // 256*40*2 = 20480
#define OFF_W1  35840                // 256*40*2 = 20480   (aliased by sT after K-loop)
#define OFF_CC  56320                // 256*4    =  1024
#define OFF_EE  57344                // 4
#define SMEM_BYTES 57348

// workspace layout: accumulators (6401 f32, zeroed each call) then f16 W^T
#define ACC_FLOATS   (ERR_DIM * CAND + CAND + 1)   // 6401
#define W16T_OFF_B   25728                         // 6432*4, 16B aligned

__global__ void __launch_bounds__(256) zero_ws_kernel(float* p, int n) {
    int i = blockIdx.x * 256 + threadIdx.x;
    if (i < n) p[i] = 0.0f;
}

// v_cvt_pk_rtz_f16_f32 -> our v2h type (builtin returns __fp16 vector; bit-cast)
static __device__ __forceinline__ v2h pkrtz(float a, float b) {
    return __builtin_bit_cast(v2h, __builtin_amdgcn_cvt_pkrtz(a, b));
}

// one-time (per launch) W -> f16 W^T [CAND][KPAD], zero-padded K
__global__ void __launch_bounds__(256) convert_w_kernel(
    const float* __restrict__ W, _Float16* __restrict__ W16T) {
    int idx = blockIdx.x * 256 + threadIdx.x;          // pair index
    int c = idx / (KPAD / 2), k = (idx % (KPAD / 2)) * 2;
    float a = (k     < IN_DIM) ? W[(size_t)k * CAND + c]       : 0.0f;
    float b = (k + 1 < IN_DIM) ? W[(size_t)(k + 1) * CAND + c] : 0.0f;
    *(v2h*)&W16T[c * KPAD + k] = pkrtz(a, b);
}

// contiguous 16-half fragment (B fragments from transposed LDS)
static __device__ __forceinline__ v16h frag16(const _Float16* p) {
    v8h lo = *(const v8h*)(p);
    v8h hi = *(const v8h*)(p + 8);
    return __builtin_shufflevector(lo, hi, 0,1,2,3,4,5,6,7,8,9,10,11,12,13,14,15);
}
// gapped fragment: elems 0..7 at p, elems 8..15 at p+16 (A-matrix K layout)
static __device__ __forceinline__ v16h frag_gap(const _Float16* p) {
    v8h lo = *(const v8h*)(p);
    v8h hi = *(const v8h*)(p + 16);
    return __builtin_shufflevector(lo, hi, 0,1,2,3,4,5,6,7,8,9,10,11,12,13,14,15);
}

// TDM: DMA one 32x256-half W^T chunk into LDS, padding rows 32 -> 40 halves.
// D# packing per CDNA5 ISA.
static __device__ __forceinline__ void tdm_w_chunk(
    const _Float16* w16t, int chunk, unsigned lds_off) {
    unsigned long long ga = (unsigned long long)(const void*)(w16t + chunk * 32);
    v4u g0;
    g0[0] = 1u;                                            // count=1 valid descriptor
    g0[1] = lds_off;                                       // lds_addr (bytes)
    g0[2] = (unsigned)(ga & 0xFFFFFFFFu);                  // global_addr[31:0]
    g0[3] = (unsigned)((ga >> 32) & 0x1FFFFFFu)            // global_addr[56:32]
          | (2u << 30);                                    // type = 2 ("image")
    v8i g1;
    g1[0] = (1 << 16)                                      // data_size = 2 bytes
          | (1 << 20)                                      // pad_enable (LDS dest)
          | (3 << 22)                                      // pad_interval: 16 dwords (64B row)
          | (3 << 25);                                     // pad_amount: 4 dwords (8 halves)
    g1[1] = (KPAD << 16);                                  // tensor_dim0[15:0] = 192
    g1[2] = (CAND << 16);                                  // tensor_dim1[15:0] = 256
    g1[3] = (32 << 16);                                    // tile_dim0 = 32
    g1[4] = CAND;                                          // tile_dim1 = 256, tile_dim2 = 0
    g1[5] = KPAD;                                          // tensor_dim0_stride[31:0] = 192
    g1[6] = 0;
    g1[7] = 0;
    v4i z4 = {0, 0, 0, 0};                                 // groups 2/3 unused (2D)
    v8i z8 = {0, 0, 0, 0, 0, 0, 0, 0};
    __builtin_amdgcn_tensor_load_to_lds(g0, g1, z4, z4, z8, 0);
}

// Fused: S = sigmoid(X W + b); ec += E^T S; cc += colsum(S*S); ee += sum(E*E)
__global__ void __launch_bounds__(256) scn_fused_kernel(
    const float* __restrict__ X,        // [N, IN_DIM]
    const float* __restrict__ E,        // [N, ERR_DIM]
    const _Float16* __restrict__ W16T,  // [CAND, KPAD] f16 (pre-transposed)
    const float* __restrict__ Bc,       // [1, CAND]
    float* __restrict__ ws_ec,          // [ERR_DIM, CAND]
    float* __restrict__ ws_cc,          // [CAND]
    float* __restrict__ ws_ee)          // [1]
{
    __shared__ __align__(16) unsigned char smem[SMEM_BYTES];
    _Float16* x_lds  = (_Float16*)(smem + OFF_X);
    _Float16* e_lds  = (_Float16*)(smem + OFF_E);
    _Float16* wT0    = (_Float16*)(smem + OFF_W0);
    _Float16* wT1    = (_Float16*)(smem + OFF_W1);
    _Float16* sT     = (_Float16*)(smem + OFF_W1);  // alias: wT1 dead after K-loop
    float*    cc_red = (float*)(smem + OFF_CC);
    float*    ee_red = (float*)(smem + OFF_EE);
    _Float16* bufs[2] = { wT0, wT1 };
    const unsigned lds_base = (unsigned)(unsigned long long)(void*)smem; // LDS aperture: low 32 bits = LDS byte address

    const int tid  = threadIdx.x;
    const int wv   = tid >> 5;        // wave id 0..7 (wave32)
    const int lane = tid & 31;
    const int n0   = blockIdx.x * TM; // N is an exact multiple of TM

    // kick off the first two W-chunk DMAs immediately (overlap with X/E staging)
    if (wv == 0) {
        tdm_w_chunk(W16T, 0, lds_base + OFF_W0);
        tdm_w_chunk(W16T, 1, lds_base + OFF_W1);
    }

    cc_red[tid] = 0.0f;
    if (tid == 0) *ee_red = 0.0f;

    // ---- stage X tile as f16 quads (float4 load -> 2x v_cvt_pk -> b64 LDS store) ----
#pragma unroll
    for (int p = 0; p < (TM * KPAD) / 4 / 256; ++p) {
        int idx = p * 256 + tid;              // quad index
        int r = idx / (KPAD / 4), k = (idx % (KPAD / 4)) * 4;
        v2h h0, h1;
        if (k < IN_DIM) {
            float4 v = *(const float4*)&X[(size_t)(n0 + r) * IN_DIM + k];
            h0 = pkrtz(v.x, v.y);
            h1 = pkrtz(v.z, v.w);
        } else {
            h0 = pkrtz(0.0f, 0.0f);
            h1 = h0;
        }
        *(v4h*)&x_lds[r * XSTR + k] = __builtin_shufflevector(h0, h1, 0, 1, 2, 3);
    }
    // ---- stage E^T tile (e_lds[d][row]) as f16 pairs; accumulate ee on f32 values ----
    float ee_loc = 0.0f;
#pragma unroll
    for (int p = 0; p < (DPAD * TM) / 2 / 256; ++p) {
        int idx = p * 256 + tid;
        int d = idx / (TM / 2), k = (idx % (TM / 2)) * 2;
        float e0 = 0.0f, e1 = 0.0f;
        if (d < ERR_DIM) {
            e0 = E[(size_t)(n0 + k) * ERR_DIM + d];
            e1 = E[(size_t)(n0 + k + 1) * ERR_DIM + d];
        }
        ee_loc += e0 * e0 + e1 * e1;
        *(v2h*)&e_lds[d * ESTR + k] = pkrtz(e0, e1);
    }
    atomicAdd(ee_red, ee_loc);

    if (wv == 0) __builtin_amdgcn_s_wait_tensorcnt(1);   // chunk 0 landed
    __syncthreads();

    // ---- wave -> tile mapping ----
    const int rt  = wv & 1;            // row tile (0/1) of the 32-row block
    const int ctg = wv >> 1;           // 4 column tiles per wave
    const int m     = lane & 15;
    const int hi    = lane >> 4;       // lanes 16-31 hold the other K-halves
    const int arow  = rt * 16 + m;     // A-fragment row (M)
    const int aoff  = hi * 8;          // A K sub-offset (ISA 16-bit A layout)
    const int bcol  = lane & 15;       // B/C/D column within 16-wide tile
    const int bkoff = hi * 16;         // B K offset (contiguous k order; unused)

    v8f acc[4];
#pragma unroll
    for (int j = 0; j < 4; ++j)
#pragma unroll
        for (int rr = 0; rr < 8; ++rr) acc[j][rr] = 0.0f;

    // ---- GEMM1: TDM double-buffered K loop ----
#pragma unroll
    for (int i = 0; i < NCHUNK; ++i) {
        const _Float16* wb = bufs[i & 1];
        v16h a = frag_gap(&x_lds[arow * XSTR + i * 32 + aoff]);
#pragma unroll
        for (int j = 0; j < 4; ++j) {
            const int c = (ctg * 4 + j) * 16 + bcol;
            v16h b = frag_gap(&wb[c * WTSTR + aoff]);
            acc[j] = __builtin_amdgcn_wmma_f32_16x16x32_f16(
                false, a, false, b, (short)0, acc[j], false, false);
        }
        __syncthreads();                       // all waves done reading buf[i&1]
        if (i + 2 < NCHUNK && wv == 0)
            tdm_w_chunk(W16T, i + 2, lds_base + ((i & 1) ? OFF_W1 : OFF_W0));
        if (i + 1 < NCHUNK) {
            if (wv == 0) {
                if (i + 2 < NCHUNK) __builtin_amdgcn_s_wait_tensorcnt(1);
                else                __builtin_amdgcn_s_wait_tensorcnt(0);
            }
            __syncthreads();                   // buf[(i+1)&1] ready for all waves
        }
    }

    // ---- epilogue: bias + sigmoid; cc partials; S^T -> LDS (one b128 store/tile) ----
#pragma unroll
    for (int j = 0; j < 4; ++j) {
        const int c = (ctg * 4 + j) * 16 + bcol;
        const float bias = Bc[c];
        float sf[8];
        float ccp = 0.0f;
#pragma unroll
        for (int r = 0; r < 8; ++r) {
            float s = 1.0f / (1.0f + __expf(-(acc[j][r] + bias)));
            sf[r] = s;
            ccp += s * s;
        }
        v8h sv;
#pragma unroll
        for (int r = 0; r < 8; r += 2) {
            v2h pk = pkrtz(sf[r], sf[r + 1]);
            sv[r] = pk[0]; sv[r + 1] = pk[1];
        }
        *(v8h*)&sT[c * SSTR + rt * 16 + hi * 8] = sv;   // rows contiguous per column
        atomicAdd(&cc_red[c], ccp);
    }
    __syncthreads();

    // ---- GEMM2: ec_partial = E^T (DPADxTM) @ S (TM x 256), K=32 in one WMMA/tile ----
    const int mt = rt;                 // M-tile over error dims
    v16h a2 = frag_gap(&e_lds[(mt * 16 + m) * ESTR + aoff]);
#pragma unroll
    for (int j = 0; j < 4; ++j) {
        const int c = (ctg * 4 + j) * 16 + bcol;
        v16h b2 = frag_gap(&sT[c * SSTR + aoff]);
        v8f z;
#pragma unroll
        for (int rr = 0; rr < 8; ++rr) z[rr] = 0.0f;
        v8f ecf = __builtin_amdgcn_wmma_f32_16x16x32_f16(
            false, a2, false, b2, (short)0, z, false, false);
#pragma unroll
        for (int r = 0; r < 8; ++r) {
            const int d = mt * 16 + (hi ? 8 + r : r);
            if (d < ERR_DIM) atomicAdd(&ws_ec[d * CAND + c], ecf[r]);
        }
    }

    __syncthreads();
    atomicAdd(&ws_cc[tid], cc_red[tid]);
    if (tid == 0) atomicAdd(ws_ee, *ee_red);
}

__global__ void __launch_bounds__(256) scn_finalize_kernel(
    const float* __restrict__ ws_ec, const float* __restrict__ ws_cc,
    const float* __restrict__ ws_ee, const float* __restrict__ r,
    float* __restrict__ out)
{
    const int c = threadIdx.x;      // 256 candidates, 1 block
    float left = 0.0f;
#pragma unroll
    for (int d = 0; d < ERR_DIM; ++d) {
        float v = ws_ec[d * CAND + c];
        left += v * v;
    }
    left /= ws_cc[c];
    const float ee = *ws_ee;
#pragma unroll
    for (int i = 0; i < NR; ++i)
        out[c * NR + i] = left - (1.0f - r[i]) * ee;
}

extern "C" void kernel_launch(void* const* d_in, const int* in_sizes, int n_in,
                              void* d_out, int out_size, void* d_ws, size_t ws_size,
                              hipStream_t stream) {
    const float* X  = (const float*)d_in[0];   // [N, IN_DIM]
    const float* E  = (const float*)d_in[1];   // [N, ERR_DIM]
    const float* W  = (const float*)d_in[2];   // [IN_DIM, CAND]
    const float* Bc = (const float*)d_in[3];   // [1, CAND]
    const float* r  = (const float*)d_in[4];   // [NR]

    float* ws       = (float*)d_ws;
    float* ws_ec    = ws;                      // ERR_DIM*CAND
    float* ws_cc    = ws + ERR_DIM * CAND;     // CAND
    float* ws_ee    = ws_cc + CAND;            // 1
    _Float16* W16T  = (_Float16*)((char*)d_ws + W16T_OFF_B);  // [CAND][KPAD] f16

    zero_ws_kernel<<<(ACC_FLOATS + 255) / 256, 256, 0, stream>>>(ws, ACC_FLOATS);
    convert_w_kernel<<<(CAND * KPAD / 2) / 256, 256, 0, stream>>>(W, W16T);
    scn_fused_kernel<<<N_ROWS / TM, 256, 0, stream>>>(X, E, W16T, Bc, ws_ec, ws_cc, ws_ee);
    scn_finalize_kernel<<<1, 256, 0, stream>>>(ws_ec, ws_cc, ws_ee, r, (float*)d_out);
}
